// MultiHeadAttention_69045894250603
// MI455X (gfx1250) — matrix operand order repaired and checked by reading, not compiled, not run on hardware
//
#include <hip/hip_runtime.h>

#define SLEN   2048
#define NH     16
#define DHD    64
#define DMODEL 1024

typedef __attribute__((ext_vector_type(16))) __bf16         bf16x16;
typedef __attribute__((ext_vector_type(8)))  float          v8f;
typedef __attribute__((ext_vector_type(8)))  unsigned short u16x8;
typedef __attribute__((ext_vector_type(4)))  unsigned short u16x4;

union U16x16 { bf16x16 v; u16x8 h[2]; };
static_assert(sizeof(U16x16) == 32, "U16x16 must be 32 bytes");

#if __has_builtin(__builtin_amdgcn_global_load_async_to_lds_b128) && \
    __has_builtin(__builtin_amdgcn_s_wait_asynccnt)
#define HAVE_ASYNC_LDS 1
// Builtin signature (from hipcc diagnostics): param0 = int4 addrspace(1)*,
// param1 = int4 addrspace(3)*, then two int immediates (offset, cpol).
typedef int gv4i __attribute__((__vector_size__(16)));
#define ASYNC_CP16(gsrc, ldst)                                      \
  __builtin_amdgcn_global_load_async_to_lds_b128(                   \
      (__attribute__((address_space(1))) gv4i*)(gsrc),              \
      (__attribute__((address_space(3))) gv4i*)(ldst), 0, 0)
#else
#define HAVE_ASYNC_LDS 0
#endif

// native f32 -> bf16 (backend emits the single-instruction convert)
__device__ __forceinline__ unsigned short f2bf(float f) {
  __bf16 b = (__bf16)f;
  return __builtin_bit_cast(unsigned short, b);
}

__device__ __forceinline__ v8f wmma_bf16(const U16x16& a, const U16x16& b, v8f c) {
  return __builtin_amdgcn_wmma_f32_16x16x32_bf16(false, a.v, false, b.v,
                                                 (short)0, c, false, false);
}

// ---------------------------------------------------------------------------
// Kernel 0: elementwise f32 -> bf16 convert (for Wo)
// ---------------------------------------------------------------------------
__global__ __launch_bounds__(256) void k_cvt_bf16(const float* __restrict__ src,
                                                  unsigned short* __restrict__ dst,
                                                  int n) {
  int base = (blockIdx.x * 256 + threadIdx.x) * 4;
  if (base + 3 < n) {
    float4 f = *(const float4*)(src + base);
    u16x4 o;
    o[0] = f2bf(f.x); o[1] = f2bf(f.y); o[2] = f2bf(f.z); o[3] = f2bf(f.w);
    *(u16x4*)(dst + base) = o;
  }
}

// ---------------------------------------------------------------------------
// Kernel 1: fused per-head q/k/v projections.
// grid = NH * (SLEN/128); block = 256 (8 waves); wave w owns a 16-row M tile.
// ---------------------------------------------------------------------------
__global__ __launch_bounds__(256) void k_proj(
    const float* __restrict__ Q, const float* __restrict__ Kin, const float* __restrict__ V,
    const float* __restrict__ Wq, const float* __restrict__ bq,
    const float* __restrict__ Wk, const float* __restrict__ bk,
    const float* __restrict__ Wv, const float* __restrict__ bv,
    unsigned short* __restrict__ qb, unsigned short* __restrict__ kb,
    unsigned short* __restrict__ vb) {
  __shared__ __align__(16) unsigned short lw[3 * 64 * 64];   // 24KB bf16 weights (B-matrices)
  __shared__ float lbias[3 * 64];
  __shared__ __align__(16) unsigned short xt[128 * 64];      // 16KB bf16 input tile
  __shared__ __align__(16) unsigned short ot[8 * 16 * 64];   // 16KB bf16 output staging

  const int tid  = threadIdx.x;
  const int h    = blockIdx.x >> 4;
  const int s0   = (blockIdx.x & 15) * 128;
  const int w    = tid >> 5, lane = tid & 31;
  const int half = lane >> 4, lr = lane & 15;
  const int m0   = w * 16;

  // stage the three 64x64 weight matrices for this head (bf16) + biases
  for (int i = tid * 4; i < 3 * 4096; i += 1024) {
    const int p = i >> 12, j = i & 4095;
    const float* Wp = (p == 0) ? Wq : (p == 1) ? Wk : Wv;
    float4 f = *(const float4*)&Wp[(size_t)h * 4096 + j];
    u16x4 o;
    o[0] = f2bf(f.x); o[1] = f2bf(f.y); o[2] = f2bf(f.z); o[3] = f2bf(f.w);
    *(u16x4*)(lw + i) = o;
  }
  if (tid < 192) {
    const int p = tid >> 6, j = tid & 63;
    const float* bp = (p == 0) ? bq : (p == 1) ? bk : bv;
    lbias[tid] = bp[(size_t)h * 64 + j];
  }

  for (int p = 0; p < 3; ++p) {
    __syncthreads();                                   // covers weight staging + xt reuse
    const float* X = (p == 0) ? Q : (p == 1) ? Kin : V;
    for (int i = tid * 4; i < 128 * 64; i += 1024) {
      const int r = i >> 6, c = i & 63;
      float4 f = *(const float4*)&X[(size_t)(s0 + r) * DMODEL + h * 64 + c];
      u16x4 o;
      o[0] = f2bf(f.x); o[1] = f2bf(f.y); o[2] = f2bf(f.z); o[3] = f2bf(f.w);
      *(u16x4*)(xt + i) = o;
    }
    __syncthreads();

    v8f acc[4] = {};
#pragma unroll
    for (int kk = 0; kk < 2; ++kk) {                   // K = 0..31, 32..63
      U16x16 a;
      const u16x8* ar = (const u16x8*)&xt[(m0 + lr) * 64 + kk * 32 + half * 16];
      a.h[0] = ar[0];                                  // K = kk*32 + half*16 + 0..7
      a.h[1] = ar[1];                                  // K = kk*32 + half*16 + 8..15
#pragma unroll
      for (int nt = 0; nt < 4; ++nt) {
        U16x16 b;
        const u16x8* br = (const u16x8*)&lw[p * 4096 + (nt * 16 + lr) * 64 + kk * 32 + half * 16];
        b.h[0] = br[0]; b.h[1] = br[1];                // col = nt*16+lr, K = kk*32+half*16 + 0..15
        acc[nt] = wmma_bf16(a, b, acc[nt]);
      }
    }

    // bias add + per-wave staging (D layout: col = lr, row = r + half*8)
    unsigned short* om = &ot[w * 16 * 64];
#pragma unroll
    for (int nt = 0; nt < 4; ++nt) {
      const float bvv = lbias[p * 64 + nt * 16 + lr];
#pragma unroll
      for (int r = 0; r < 8; ++r)
        om[(r + half * 8) * 64 + nt * 16 + lr] = f2bf(acc[nt][r] + bvv);
    }
    // coalesced store: each lane writes half of one row (32 bf16 = 64B)
    unsigned short* dst = (p == 0) ? qb : (p == 1) ? kb : vb;
    const int row = lane >> 1, ch = (lane & 1) * 32;
    const u16x8* sv = (const u16x8*)&om[row * 64 + ch];
    u16x8* dv = (u16x8*)&dst[((size_t)h * SLEN + s0 + m0 + row) * 64 + ch];
    dv[0] = sv[0]; dv[1] = sv[1]; dv[2] = sv[2]; dv[3] = sv[3];
  }
}

// ---------------------------------------------------------------------------
// Kernel 2: flash attention per head. grid = NH * (SLEN/128); block = 256.
// Wave w owns 16 query rows; 128-key blocks stream through LDS.
// K tile is staged with GLOBAL_LOAD_ASYNC_TO_LDS_B128 when available.
// ---------------------------------------------------------------------------
__global__ __launch_bounds__(256) void k_attn(
    const unsigned short* __restrict__ qb, const unsigned short* __restrict__ kb,
    const unsigned short* __restrict__ vb, unsigned short* __restrict__ ob) {
  __shared__ __align__(16) unsigned short kt[128 * 64];    // 16KB  K tile [key][d]
  __shared__ __align__(16) unsigned short vt[64 * 136];    // 17KB  V tile transposed [e][key]
  __shared__ __align__(16) unsigned short pt[8 * 16 * 64]; // 16KB  per-wave P / out staging

  const int tid  = threadIdx.x;
  const int h    = blockIdx.x >> 4;
  const int sq0  = (blockIdx.x & 15) * 128;
  const int w    = tid >> 5, lane = tid & 31;
  const int half = lane >> 4, lr = lane & 15;

  // persistent q A-fragments (16x64 -> two K=32 fragments)
  U16x16 aq[2];
  {
    const int row = sq0 + w * 16 + lr;
    const u16x8* qr = (const u16x8*)(qb + ((size_t)h * SLEN + row) * 64);
#pragma unroll
    for (int kk = 0; kk < 2; ++kk) {
      aq[kk].h[0] = qr[kk * 4 + half * 2];
      aq[kk].h[1] = qr[kk * 4 + half * 2 + 1];
    }
  }

  v8f O[4] = {};
  float M[8], L[8];
#pragma unroll
  for (int r = 0; r < 8; ++r) { M[r] = -1e30f; L[r] = 0.0f; }

  for (int kb0 = 0; kb0 < SLEN; kb0 += 128) {
    __syncthreads();
    // --- K tile: flat 16KB copy ---
#if HAVE_ASYNC_LDS
    {
      const unsigned short* src = kb + ((size_t)h * SLEN + kb0) * 64;
      for (int i = tid * 8; i < 128 * 64; i += 2048)
        ASYNC_CP16(src + i, kt + i);
    }
#else
    {
      const u16x8* src = (const u16x8*)(kb + ((size_t)h * SLEN + kb0) * 64);
      u16x8* dstv = (u16x8*)kt;
      for (int i = tid; i < 1024; i += 256) dstv[i] = src[i];
    }
#endif
    // --- V tile: transpose into [e][key] ---
    {
      const unsigned short* src = vb + ((size_t)h * SLEN + kb0) * 64;
      for (int i = tid * 8; i < 128 * 64; i += 2048) {
        const int r = i >> 6, c = i & 63;
        u16x8 vv = *(const u16x8*)(src + i);
#pragma unroll
        for (int j = 0; j < 8; ++j) vt[(c + j) * 136 + r] = vv[j];
      }
    }
#if HAVE_ASYNC_LDS
    __builtin_amdgcn_s_wait_asynccnt(0);
#endif
    __syncthreads();

    // scores: 8 N-tiles of 16 keys, K=64 contraction (2 WMMA each)
    v8f st[8];
#pragma unroll
    for (int nt = 0; nt < 8; ++nt) {
      v8f acc = {};
#pragma unroll
      for (int kk = 0; kk < 2; ++kk) {
        U16x16 b;
        const u16x8* br = (const u16x8*)&kt[(nt * 16 + lr) * 64 + kk * 32 + half * 16];
        b.h[0] = br[0]; b.h[1] = br[1];
        acc = wmma_bf16(aq[kk], b, acc);
      }
#pragma unroll
      for (int i = 0; i < 8; ++i) acc[i] *= 0.125f;    // 1/sqrt(64)
      st[nt] = acc;
    }

    // online softmax: row stats via intra-16-lane xor reductions
    float nm[8], al[8];
#pragma unroll
    for (int r = 0; r < 8; ++r) {
      float mx = st[0][r];
#pragma unroll
      for (int nt = 1; nt < 8; ++nt) mx = fmaxf(mx, st[nt][r]);
#pragma unroll
      for (int d = 1; d < 16; d <<= 1) mx = fmaxf(mx, __shfl_xor(mx, d, 32));
      nm[r] = fmaxf(M[r], mx);
      al[r] = __expf(M[r] - nm[r]);
      M[r]  = nm[r];
    }
    float rs[8];
#pragma unroll
    for (int r = 0; r < 8; ++r) rs[r] = 0.0f;
#pragma unroll
    for (int nt = 0; nt < 8; ++nt)
#pragma unroll
      for (int r = 0; r < 8; ++r) {
        const float pv = __expf(st[nt][r] - nm[r]);
        st[nt][r] = pv;
        rs[r] += pv;
      }
#pragma unroll
    for (int r = 0; r < 8; ++r) {
      float t = rs[r];
#pragma unroll
      for (int d = 1; d < 16; d <<= 1) t += __shfl_xor(t, d, 32);
      L[r] = L[r] * al[r] + t;
    }
#pragma unroll
    for (int nv = 0; nv < 4; ++nv)
#pragma unroll
      for (int r = 0; r < 8; ++r) O[nv][r] *= al[r];

    // P @ V: two 64-key halves, P goes through per-wave LDS to become A-fragments
    unsigned short* pw = &pt[w * 16 * 64];
#pragma unroll
    for (int hk = 0; hk < 2; ++hk) {
#pragma unroll
      for (int nt2 = 0; nt2 < 4; ++nt2)
#pragma unroll
        for (int r = 0; r < 8; ++r)
          pw[(r + half * 8) * 64 + nt2 * 16 + lr] = f2bf(st[hk * 4 + nt2][r]);
      // same-wave DS ops are in-order; no cross-wave sharing here
#pragma unroll
      for (int kk2 = 0; kk2 < 2; ++kk2) {
        U16x16 ap;
        const u16x8* pr = (const u16x8*)&pw[lr * 64 + kk2 * 32 + half * 16];
        ap.h[0] = pr[0]; ap.h[1] = pr[1];
#pragma unroll
        for (int nv = 0; nv < 4; ++nv) {
          U16x16 b;
          const u16x8* br =
              (const u16x8*)&vt[(nv * 16 + lr) * 136 + hk * 64 + kk2 * 32 + half * 16];
          b.h[0] = br[0]; b.h[1] = br[1];
          O[nv] = wmma_bf16(ap, b, O[nv]);
        }
      }
    }
  }

  // normalize and store o[s][h*64+e] (bf16) via per-wave LDS transpose
  unsigned short* pw = &pt[w * 16 * 64];
#pragma unroll
  for (int nv = 0; nv < 4; ++nv)
#pragma unroll
    for (int r = 0; r < 8; ++r) {
      const float inv = 1.0f / L[r];
      pw[(r + half * 8) * 64 + nv * 16 + lr] = f2bf(O[nv][r] * inv);
    }
  const int row = lane >> 1, ch = (lane & 1) * 32;
  const u16x8* sv = (const u16x8*)&pw[row * 64 + ch];
  u16x8* dv = (u16x8*)&ob[(size_t)(sq0 + w * 16 + row) * DMODEL + h * 64 + ch];
  dv[0] = sv[0]; dv[1] = sv[1]; dv[2] = sv[2]; dv[3] = sv[3];
}

// ---------------------------------------------------------------------------
// Kernel 3: output projection: out = concat(2048x1024) @ Wo^T + bo (f32 out).
// grid = (SLEN/128)*(DMODEL/128); block = 256; 128x128 C tile per workgroup.
// A/B tiles staged with GLOBAL_LOAD_ASYNC_TO_LDS_B128 when available.
// ---------------------------------------------------------------------------
__global__ __launch_bounds__(256) void k_oproj(
    const unsigned short* __restrict__ ob, const unsigned short* __restrict__ wob,
    const float* __restrict__ bo, float* __restrict__ out) {
  __shared__ __align__(16) unsigned short smem[2 * 128 * 64];  // 32KB: A tile + B tile
  unsigned short* at = smem;
  unsigned short* bt = smem + 128 * 64;

  const int tid  = threadIdx.x;
  const int s0   = (blockIdx.x >> 3) * 128;
  const int o0   = (blockIdx.x & 7) * 128;
  const int w    = tid >> 5, lane = tid & 31;
  const int half = lane >> 4, lr = lane & 15;
  const int m0   = w * 16;

  v8f acc[8] = {};
  for (int kk0 = 0; kk0 < DMODEL; kk0 += 64) {
    __syncthreads();
#if HAVE_ASYNC_LDS
    for (int i = tid * 8; i < 128 * 64; i += 2048) {
      const int r = i >> 6, c = i & 63;
      ASYNC_CP16(ob + (size_t)(s0 + r) * DMODEL + kk0 + c, at + i);
      ASYNC_CP16(wob + (size_t)(o0 + r) * DMODEL + kk0 + c, bt + i);
    }
    __builtin_amdgcn_s_wait_asynccnt(0);
#else
    for (int i = tid * 8; i < 128 * 64; i += 2048) {
      const int r = i >> 6, c = i & 63;
      *(u16x8*)(at + i) = *(const u16x8*)(ob + (size_t)(s0 + r) * DMODEL + kk0 + c);
      *(u16x8*)(bt + i) = *(const u16x8*)(wob + (size_t)(o0 + r) * DMODEL + kk0 + c);
    }
#endif
    if (kk0 + 64 < DMODEL) {   // hint next K-slab toward cache (global_prefetch_b8)
      __builtin_prefetch(&ob[(size_t)(s0 + (tid >> 1)) * DMODEL + kk0 + 64], 0, 1);
      __builtin_prefetch(&wob[(size_t)(o0 + (tid >> 1)) * DMODEL + kk0 + 64], 0, 1);
    }
    __syncthreads();
#pragma unroll
    for (int kk = 0; kk < 2; ++kk) {
      U16x16 a;
      const u16x8* ar = (const u16x8*)&at[(m0 + lr) * 64 + kk * 32 + half * 16];
      a.h[0] = ar[0]; a.h[1] = ar[1];
#pragma unroll
      for (int nt = 0; nt < 8; ++nt) {
        U16x16 b;
        const u16x8* br = (const u16x8*)&bt[(nt * 16 + lr) * 64 + kk * 32 + half * 16];
        b.h[0] = br[0]; b.h[1] = br[1];
        acc[nt] = wmma_bf16(a, b, acc[nt]);
      }
    }
  }

  __syncthreads();   // done reading at/bt everywhere; reuse as f32 staging
  float* sw = (float*)smem + w * 16 * 64;   // 4KB per wave
#pragma unroll
  for (int hh = 0; hh < 2; ++hh) {
#pragma unroll
    for (int nt2 = 0; nt2 < 4; ++nt2) {
      const int nt = hh * 4 + nt2;
      const float bb = bo[o0 + nt * 16 + lr];
#pragma unroll
      for (int r = 0; r < 8; ++r)
        sw[(r + half * 8) * 64 + nt2 * 16 + lr] = acc[nt][r] + bb;
    }
    const int row = lane >> 1, ch = (lane & 1) * 32;
    const float4* sv = (const float4*)&sw[row * 64 + ch];
    float4* dv = (float4*)&out[(size_t)(s0 + m0 + row) * DMODEL + o0 + hh * 64 + ch];
#pragma unroll
    for (int q = 0; q < 8; ++q) dv[q] = sv[q];
  }
}

// ---------------------------------------------------------------------------
extern "C" void kernel_launch(void* const* d_in, const int* in_sizes, int n_in,
                              void* d_out, int out_size, void* d_ws, size_t ws_size,
                              hipStream_t stream) {
  const float* Q   = (const float*)d_in[0];
  const float* Kin = (const float*)d_in[1];
  const float* V   = (const float*)d_in[2];
  const float* Wq  = (const float*)d_in[3];
  const float* bq  = (const float*)d_in[4];
  const float* Wk  = (const float*)d_in[5];
  const float* bk  = (const float*)d_in[6];
  const float* Wv  = (const float*)d_in[7];
  const float* bv  = (const float*)d_in[8];
  const float* Wo  = (const float*)d_in[9];
  const float* bo  = (const float*)d_in[10];

  const size_t nph = (size_t)NH * SLEN * DHD;          // 2M elems per projection buffer
  unsigned short* qb  = (unsigned short*)d_ws;
  unsigned short* kb  = qb + nph;
  unsigned short* vb  = kb + nph;
  unsigned short* ob  = vb + nph;                      // [S][H*DHD] bf16
  unsigned short* wob = ob + (size_t)SLEN * DMODEL;    // [DMODEL][DMODEL] bf16

  k_cvt_bf16<<<(DMODEL * DMODEL) / 1024, 256, 0, stream>>>(Wo, wob, DMODEL * DMODEL);
  k_proj<<<NH * (SLEN / 128), 256, 0, stream>>>(Q, Kin, V, Wq, bq, Wk, bk, Wv, bv,
                                                qb, kb, vb);
  k_attn<<<NH * (SLEN / 128), 256, 0, stream>>>(qb, kb, vb, ob);
  k_oproj<<<(SLEN / 128) * (DMODEL / 128), 256, 0, stream>>>(ob, wob, bo, (float*)d_out);
}
